// GWNNLayer_68341519613981
// MI455X (gfx1250) — matrix operand order repaired and checked by reading, not compiled, not run on hardware
//
#include <hip/hip_runtime.h>

// GWNN layer on MI455X (gfx1250).
//   out = (wavelets*filt[None,:]) @ (wavelets_inv @ (features @ W))
// Roofline: 2x 268MB fp32 streamed from HBM => ~23us floor @ 23.3 TB/s;
// bf16 WMMA (fp32 acc) keeps math well under that; fp32 WMMA (K=4) would not.
// filt is folded into rows of the intermediate S (diag commutes through).
//
// CDNA5 paths used: v_wmma_f32_16x16x32_bf16, global_load_async_to_lds_b128
// (ASYNCcnt double-buffered B staging), s_wait_asynccnt, global_prefetch_b8.

typedef __attribute__((ext_vector_type(16))) __bf16 v16bf;
typedef __attribute__((ext_vector_type(8)))  __bf16 v8bf;
typedef __attribute__((ext_vector_type(8)))  float  v8f;
typedef __attribute__((ext_vector_type(4)))  float  v4f;

#define NROWS  8192
#define IN_CH  128
#define OUT_CH 64

#define KBLKS   (NROWS / 32)      // 256 K-steps of 32 (bf16 WMMA K)
#define QSTEPS  4                 // K-steps per LDS chunk
#define NCHUNK  (KBLKS / QSTEPS)  // 64 chunks of K=128
#define CHUNK_BYTES (OUT_CH * 128 * 2)  // 16 KB of bf16 B data per chunk
#define RUN_PAD 80                // 64B column-run padded to 80B (bank-conflict-free)
#define LDSBUF_BYTES (QSTEPS * OUT_CH * RUN_PAD)  // 20480 B per buffer

// ---------------------------------------------------------------------------
// Stage 1: T = features @ W -> bf16 in B-fragment layout:
//   Tprep[((k>>5)*64 + col)*32 + (k&31)] = bf16(T[k,col])
// ---------------------------------------------------------------------------
__global__ void __launch_bounds__(256)
prep_T_kernel(const float* __restrict__ features,
              const float* __restrict__ W,
              __bf16* __restrict__ Tprep)
{
    const int col = threadIdx.x;                   // 0..63
    const int row = blockIdx.x * 4 + threadIdx.y;  // 0..8191
    const float* f = features + (size_t)row * IN_CH;

    float acc = 0.0f;
    #pragma unroll 8
    for (int k = 0; k < IN_CH; ++k)
        acc = __builtin_fmaf(f[k], W[k * OUT_CH + col], acc);

    Tprep[(size_t)(((row >> 5) * OUT_CH) + col) * 32 + (row & 31)] = (__bf16)acc;
}

// ---------------------------------------------------------------------------
// Big GEMM: Out[8192,64] = A[8192,8192](fp32) @ Bprep[8192,64](bf16, B-layout)
// Block = 128 threads (4 waves); wave tile = 16(M) x 64(N); K loop in chunks
// of 128, B chunk staged in LDS via double-buffered async copies.
// ---------------------------------------------------------------------------
template <bool FUSE_FILT_BF16>
__global__ void __launch_bounds__(128)
gemm_wmma_kernel(const float*  __restrict__ A,
                 const __bf16* __restrict__ Bp,
                 const float*  __restrict__ filt,
                 __bf16*       __restrict__ OutBf,
                 float*        __restrict__ OutF)
{
    __shared__ __attribute__((aligned(128))) unsigned char smem[2][LDSBUF_BYTES];

    const int tid  = threadIdx.x;
    const int lane = tid & 31;
    const int wave = tid >> 5;
    const int n    = lane & 15;     // B col-within-16 / A row-within-16
    const int h    = lane >> 4;     // half-wave select
    const int m0   = blockIdx.x * 64 + wave * 16;
    const int mrow = m0 + n;

    v8f acc[4] = {};

    // ISA 16-bit A 16x32 layout: lane (n,h) holds row n,
    // K = {8h..8h+7} U {16+8h..16+8h+7} (two contiguous fp32 runs).
    const float* arow = A + (size_t)mrow * NROWS + 8 * h;

    // LDS base as a wave-relative byte offset (shared-aperture low 32 bits).
    const uint32_t lds0 = (uint32_t)(uintptr_t)(&smem[0][0]);
    const uint32_t lds1 = (uint32_t)(uintptr_t)(&smem[1][0]);

    // Issue one 16KB chunk: 128 threads x 8 async b128 packets (16B each).
    // LDS destination inserts 16B pad per 64B column-run: d = s + (s>>6)<<4.
    auto issue_chunk = [&](int c, uint32_t dbase) {
        const char* src = (const char*)Bp + (size_t)c * CHUNK_BYTES;
        #pragma unroll
        for (int r = 0; r < 8; ++r) {
            const int sb = (r * 128 + tid) * 16;
            const uint32_t dst = dbase + (uint32_t)(sb + ((sb >> 6) << 4));
            const unsigned long long ga =
                (unsigned long long)(uintptr_t)(src + sb);
            asm volatile("global_load_async_to_lds_b128 %0, %1, off"
                         :: "v"(dst), "v"(ga) : "memory");
        }
    };

    issue_chunk(0, lds0);

    for (int c = 0; c < NCHUNK; ++c) {
        const int cur = c & 1;
        // Prefetch chunk c+1 into the other buffer (wraps at the end so the
        // wait immediate stays constant; strays drain at s_endpgm).
        issue_chunk((c + 1) & (NCHUNK - 1), cur ? lds0 : lds1);
        // 8 just-issued outstanding; <=8 means chunk c's copies completed
        // (per-wave in-order completion of async loads).
        asm volatile("s_wait_asynccnt 0x8" ::: "memory");
        __syncthreads();   // publish every wave's slice of chunk c

        const unsigned char* bbuf = cur ? &smem[1][0] : &smem[0][0];

        #pragma unroll
        for (int q = 0; q < QSTEPS; ++q) {
            const int kb = c * QSTEPS + q;
            const float* ap = arow + (size_t)kb * 32;
            v4f f0 = *(const v4f*)(ap);
            v4f f1 = *(const v4f*)(ap + 4);
            v4f f2 = *(const v4f*)(ap + 16);
            v4f f3 = *(const v4f*)(ap + 20);
            __builtin_prefetch((const void*)(ap + 512), 0, 0);  // ~4 chunks ahead

            v16bf a;
            #pragma unroll
            for (int i = 0; i < 4; ++i) {
                a[i]      = (__bf16)f0[i];
                a[4 + i]  = (__bf16)f1[i];
                a[8 + i]  = (__bf16)f2[i];
                a[12 + i] = (__bf16)f3[i];
            }

            #pragma unroll
            for (int j = 0; j < 4; ++j) {
                // Fragment: col 16j+n, K-run half h. Padded run stride 80B
                // -> lane n starts at bank 20n mod 64: conflict-free b128s.
                const unsigned char* bptr =
                    bbuf + (q * OUT_CH + 16 * j + n) * RUN_PAD + 16 * h;
                v8bf blo = *(const v8bf*)(bptr);
                v8bf bhi = *(const v8bf*)(bptr + 32);
                v16bf b;
                #pragma unroll
                for (int i = 0; i < 8; ++i) { b[i] = blo[i]; b[8 + i] = bhi[i]; }
                acc[j] = __builtin_amdgcn_wmma_f32_16x16x32_bf16(
                             false, a, false, b, (short)0, acc[j], false, false);
            }
        }
        __syncthreads();   // all waves done reading before buffer is re-filled
    }

    // Epilogue. C/D layout: lane (n,h), VGPR r -> M = m0 + 8h + r, N = n.
    #pragma unroll
    for (int j = 0; j < 4; ++j) {
        const int col = 16 * j + n;
        #pragma unroll
        for (int r = 0; r < 8; ++r) {
            const int m = m0 + 8 * h + r;
            float v = acc[j][r];
            if (FUSE_FILT_BF16) {
                v *= filt[m];   // fold diag(filt) into rows of S
                OutBf[(size_t)(((m >> 5) * OUT_CH) + col) * 32 + (m & 31)] =
                    (__bf16)v;
            } else {
                OutF[(size_t)m * OUT_CH + col] = v;
            }
        }
    }
}

// ---------------------------------------------------------------------------
// inputs: features[8192*128], weight_matrix[128*64], filt[8192],
//         wavelets[8192*8192], wavelets_inv[8192*8192]  (all fp32)
// output: [8192,64] fp32. workspace: Tprep + Sprep (1 MB bf16 each).
// ---------------------------------------------------------------------------
extern "C" void kernel_launch(void* const* d_in, const int* in_sizes, int n_in,
                              void* d_out, int out_size, void* d_ws, size_t ws_size,
                              hipStream_t stream)
{
    const float* features     = (const float*)d_in[0];
    const float* weight       = (const float*)d_in[1];
    const float* filt         = (const float*)d_in[2];
    const float* wavelets     = (const float*)d_in[3];
    const float* wavelets_inv = (const float*)d_in[4];
    float*       out          = (float*)d_out;

    __bf16* Tprep = (__bf16*)d_ws;                   // 1 MB
    __bf16* Sprep = Tprep + (size_t)NROWS * OUT_CH;  // 1 MB

    prep_T_kernel<<<dim3(NROWS / 4), dim3(OUT_CH, 4), 0, stream>>>(
        features, weight, Tprep);

    gemm_wmma_kernel<true><<<dim3(NROWS / 64), dim3(128), 0, stream>>>(
        wavelets_inv, Tprep, filt, Sprep, nullptr);

    gemm_wmma_kernel<false><<<dim3(NROWS / 64), dim3(128), 0, stream>>>(
        wavelets, Sprep, nullptr, nullptr, out);
}
